// EdgeSelector_32607391711818
// MI455X (gfx1250) — matrix operand order repaired and checked by reading, not compiled, not run on hardware
//
#include <hip/hip_runtime.h>
#include <hip/hip_bf16.h>
#include <math.h>

typedef __attribute__((ext_vector_type(16))) _Float16 v16h;
typedef __attribute__((ext_vector_type(8)))  _Float16 v8h;
typedef __attribute__((ext_vector_type(8)))  float    v8f;
typedef __attribute__((ext_vector_type(4)))  float    v4f;

#define NODE_D 128
#define K2     256          // 2 * NODE_D
#define TILE_M 64           // 4 sub-tiles of 16 edges

union AFrag { v16h v; v8h h[2]; };

// ---------------------------------------------------------------------------
// Pack W1 (f32 [256,128] row-major, K x N) into WMMA-B fragment order, f16.
//   f = ((ntile*8 + kk)*32 + lane)*16 + j
//   K = kk*32 + (lane>=16 ? 16 : 0) + j ; N = ntile*16 + (lane & 15)
// ---------------------------------------------------------------------------
__global__ __launch_bounds__(256) void pack_w1_kernel(
    const float* __restrict__ W1, _Float16* __restrict__ w1p) {
  int f = blockIdx.x * 256 + threadIdx.x;      // 0 .. 32767
  int j    = f & 15;
  int lane = (f >> 4) & 31;
  int kk   = (f >> 9) & 7;
  int nt   = f >> 12;
  int K = kk * 32 + ((lane >= 16) ? 16 : 0) + j;
  int N = nt * 16 + (lane & 15);
  w1p[f] = (_Float16)W1[K * NODE_D + N];
}

// ---------------------------------------------------------------------------
// scores[e] = relu(concat(emb[src],emb[dst]) @ W1 + b1) @ W2 + b2
// Block = 8 waves. Wave (nh = wv&3, mh = wv>>2):
//   owns N columns 32*nh..32*nh+31 (2 B-fragment sets, register resident)
//   owns M sub-tiles {2*mh, 2*mh+1}
// Flattened 16-step WMMA loop with depth-3 rotating A-fragment buffer.
// ---------------------------------------------------------------------------
__global__ __launch_bounds__(256, 1) void edge_gemm_kernel(
    const int*   __restrict__ edge_list,   // [E][2] int32
    const float* __restrict__ emb,         // [N][128]
    const _Float16* __restrict__ w1p,      // packed B fragments
    const float* __restrict__ b1,          // [128]
    const float* __restrict__ w2,          // [128]
    const float* __restrict__ b2,          // [1]
    float*       __restrict__ scores,      // [E]
    int E, int ntiles) {
  __shared__ __align__(16) _Float16 ah[TILE_M * K2];  // 32 KB A tile (f16)
  __shared__ float ssum[TILE_M];

  const int t    = threadIdx.x;
  const int lane = t & 31;
  const int wv   = t >> 5;
  const int nh   = wv & 3;                 // N column-pair group (0..3)
  const int mh   = wv >> 2;                // M half (0..1)

  // --- B fragments for the two owned N tiles ---
  v16h bfrag0[8], bfrag1[8];
  {
    const _Float16* base0 = w1p + ((size_t)((nh * 2 + 0) * 8) * 32 + lane) * 16;
    const _Float16* base1 = w1p + ((size_t)((nh * 2 + 1) * 8) * 32 + lane) * 16;
    #pragma unroll
    for (int kk = 0; kk < 8; ++kk) {
      bfrag0[kk] = *(const v16h*)(base0 + (size_t)kk * 32 * 16);
      bfrag1[kk] = *(const v16h*)(base1 + (size_t)kk * 32 * 16);
    }
  }

  const int   ncol0 = (nh * 2 + 0) * 16 + (lane & 15);
  const int   ncol1 = (nh * 2 + 1) * 16 + (lane & 15);
  const float b1n0  = b1[ncol0];
  const float b1n1  = b1[ncol1];
  const float w2n0  = w2[ncol0];
  const float w2n1  = w2[ncol1];
  const float b2v   = b2[0];

  // A-fragment addressing (contiguous k order: k = 16*(lane>>4) + s)
  const int m_a   = lane & 15;
  const int off_a = (lane >> 4) * 16;

  // step i (0..15): sub-tile s = mh*2 + (i>>3), k-step kk = i&7
  auto aptr = [&](int i) -> const _Float16* {
    return ah + ((mh * 2 + (i >> 3)) * 16 + m_a) * K2 + (i & 7) * 32 + off_a;
  };

  // Gather: 4 threads per edge, 64 consecutive f32 (256 B) each
  const int gm = t >> 2;
  const int gi = t & 3;

  // halving-butterfly output index r for this lane
  const int r_out = 4 * (lane & 1) + 2 * ((lane >> 1) & 1) + ((lane >> 2) & 1);

  for (int tile = blockIdx.x; tile < ntiles; tile += gridDim.x) {
    if (t < TILE_M) ssum[t] = 0.0f;

    // ---- Gather A tile: 64 edges x 256 f32 -> f16 LDS ----
    {
      int e  = tile * TILE_M + gm;
      int ec = (e < E) ? e : (E - 1);
      int node = edge_list[2 * ec + (gi >> 1)];
      const float* row = emb + (size_t)node * NODE_D + (gi & 1) * 64;
      _Float16* dsth = ah + gm * K2 + gi * 64;
      #pragma unroll
      for (int q = 0; q < 16; ++q) {
        v4f f = *(const v4f*)(row + q * 4);
        dsth[q * 4 + 0] = (_Float16)f.x;
        dsth[q * 4 + 1] = (_Float16)f.y;
        dsth[q * 4 + 2] = (_Float16)f.z;
        dsth[q * 4 + 3] = (_Float16)f.w;
      }
    }
    __syncthreads();

    // ---- accumulators ----
    v8f acc0[2], acc1[2];
    #pragma unroll
    for (int si = 0; si < 2; ++si) {
      #pragma unroll
      for (int r = 0; r < 8; ++r) { acc0[si][r] = b1n0; acc1[si][r] = b1n1; }
    }

    // ---- flattened 16-step WMMA loop, depth-3 A prefetch pipeline ----
    AFrag abuf[3];
    #pragma unroll
    for (int p = 0; p < 3; ++p) {
      const _Float16* ab = aptr(p);
      abuf[p].h[0] = *(const v8h*)(ab);
      abuf[p].h[1] = *(const v8h*)(ab + 8);
    }
    #pragma unroll
    for (int i = 0; i < 16; ++i) {
      const int bi = i % 3;
      const int si = i >> 3;
      const int kk = i & 7;
      acc0[si] = __builtin_amdgcn_wmma_f32_16x16x32_f16(
          false, abuf[bi].v, false, bfrag0[kk], (short)0, acc0[si], false, false);
      acc1[si] = __builtin_amdgcn_wmma_f32_16x16x32_f16(
          false, abuf[bi].v, false, bfrag1[kk], (short)0, acc1[si], false, false);
      if (i + 3 < 16) {                      // refill this slot 3 steps ahead
        const _Float16* ab = aptr(i + 3);
        abuf[bi].h[0] = *(const v8h*)(ab);
        abuf[bi].h[1] = *(const v8h*)(ab + 8);
      }
    }

    // ---- ReLU*w2 (both N tiles pre-combined), butterfly, ds_add ----
    #pragma unroll
    for (int si = 0; si < 2; ++si) {
      float c[8];
      #pragma unroll
      for (int r = 0; r < 8; ++r)
        c[r] = fmaxf(acc0[si][r], 0.0f) * w2n0 + fmaxf(acc1[si][r], 0.0f) * w2n1;

      #pragma unroll
      for (int i = 0; i < 4; ++i) {             // xor 1: 8 -> 4
        float send = (lane & 1) ? c[i] : c[i + 4];
        float recv = __shfl_xor(send, 1);
        float keep = (lane & 1) ? c[i + 4] : c[i];
        c[i] = keep + recv;
      }
      #pragma unroll
      for (int i = 0; i < 2; ++i) {             // xor 2: 4 -> 2
        float send = (lane & 2) ? c[i] : c[i + 2];
        float recv = __shfl_xor(send, 2);
        float keep = (lane & 2) ? c[i + 2] : c[i];
        c[i] = keep + recv;
      }
      {                                          // xor 4: 2 -> 1
        float send = (lane & 4) ? c[0] : c[1];
        float recv = __shfl_xor(send, 4);
        float keep = (lane & 4) ? c[1] : c[0];
        c[0] = keep + recv;
      }
      c[0] += __shfl_xor(c[0], 8);               // full 16-lane sum

      if (!(lane & 8)) {                         // one ds_add_f32, 16 lanes
        int m = (mh * 2 + si) * 16 + ((lane >> 4) * 8) + r_out;
        atomicAdd(&ssum[m], c[0]);
      }
    }
    __syncthreads();

    if (t < TILE_M) {
      int e = tile * TILE_M + t;
      if (e < E) scores[e] = ssum[t] + b2v;
    }
  }
}

// ---------------------------------------------------------------------------
// Softmax over E scores
// ---------------------------------------------------------------------------
__global__ __launch_bounds__(256) void reduce_max_kernel(
    const float* __restrict__ s, int n, float* __restrict__ part) {
  float m = -INFINITY;
  for (int i = blockIdx.x * 256 + threadIdx.x; i < n; i += gridDim.x * 256)
    m = fmaxf(m, s[i]);
  #pragma unroll
  for (int o = 16; o >= 1; o >>= 1) m = fmaxf(m, __shfl_xor(m, o));
  __shared__ float sm[8];
  if ((threadIdx.x & 31) == 0) sm[threadIdx.x >> 5] = m;
  __syncthreads();
  if (threadIdx.x < 8) {
    float v = sm[threadIdx.x];
    v = fmaxf(v, __shfl_xor(v, 4));
    v = fmaxf(v, __shfl_xor(v, 2));
    v = fmaxf(v, __shfl_xor(v, 1));
    if (threadIdx.x == 0) part[blockIdx.x] = v;
  }
}

__global__ __launch_bounds__(256) void final_max_kernel(
    const float* __restrict__ part, int np, float* __restrict__ stat) {
  float m = -INFINITY;
  for (int i = threadIdx.x; i < np; i += 256) m = fmaxf(m, part[i]);
  #pragma unroll
  for (int o = 16; o >= 1; o >>= 1) m = fmaxf(m, __shfl_xor(m, o));
  __shared__ float sm[8];
  if ((threadIdx.x & 31) == 0) sm[threadIdx.x >> 5] = m;
  __syncthreads();
  if (threadIdx.x == 0) {
    float v = -INFINITY;
    #pragma unroll
    for (int i = 0; i < 8; ++i) v = fmaxf(v, sm[i]);
    stat[0] = v;
    stat[1] = 0.0f;
  }
}

__global__ __launch_bounds__(256) void exp_sum_kernel(
    const float* __restrict__ s, float* __restrict__ out, int n,
    float* __restrict__ stat) {
  const float mx = stat[0];
  float acc = 0.0f;
  for (int i = blockIdx.x * 256 + threadIdx.x; i < n; i += gridDim.x * 256) {
    float e = __expf(s[i] - mx);
    out[i] = e;
    acc += e;
  }
  #pragma unroll
  for (int o = 16; o >= 1; o >>= 1) acc += __shfl_xor(acc, o);
  __shared__ float sm[8];
  if ((threadIdx.x & 31) == 0) sm[threadIdx.x >> 5] = acc;
  __syncthreads();
  if (threadIdx.x == 0) {
    float v = 0.0f;
    #pragma unroll
    for (int i = 0; i < 8; ++i) v += sm[i];
    atomicAdd(&stat[1], v);
  }
}

__global__ __launch_bounds__(256) void normalize_kernel(
    float* __restrict__ out, int n, const float* __restrict__ stat) {
  const float inv = 1.0f / stat[1];
  for (int i = blockIdx.x * 256 + threadIdx.x; i < n; i += gridDim.x * 256)
    out[i] *= inv;
}

// ---------------------------------------------------------------------------
extern "C" void kernel_launch(void* const* d_in, const int* in_sizes, int n_in,
                              void* d_out, int out_size, void* d_ws, size_t ws_size,
                              hipStream_t stream) {
  const int*   edge_list = (const int*)  d_in[0];
  const float* emb       = (const float*)d_in[1];
  const float* W1        = (const float*)d_in[2];
  const float* b1        = (const float*)d_in[3];
  const float* W2        = (const float*)d_in[4];
  const float* b2        = (const float*)d_in[5];
  float* out = (float*)d_out;

  const int E      = in_sizes[0] / 2;
  const int ntiles = (E + TILE_M - 1) / TILE_M;

  char* ws = (char*)d_ws;
  size_t o_w1p  = (((size_t)E * 4) + 255) & ~(size_t)255;
  size_t o_part = o_w1p + 65536;
  size_t o_stat = o_part + 512 * sizeof(float);
  float*    scores = (float*)ws;
  _Float16* w1p    = (_Float16*)(ws + o_w1p);
  float*    part   = (float*)(ws + o_part);
  float*    stat   = (float*)(ws + o_stat);      // [0]=max, [1]=sum

  pack_w1_kernel<<<128, 256, 0, stream>>>(W1, w1p);

  int grid = ntiles < 2560 ? ntiles : 2560;
  edge_gemm_kernel<<<grid, 256, 0, stream>>>(edge_list, emb, w1p, b1, W2, b2,
                                             scores, E, ntiles);

  const int P = 512;
  reduce_max_kernel<<<P, 256, 0, stream>>>(scores, E, part);
  final_max_kernel<<<1, 256, 0, stream>>>(part, P, stat);
  exp_sum_kernel<<<P, 256, 0, stream>>>(scores, out, E, stat);
  normalize_kernel<<<P, 256, 0, stream>>>(out, E, stat);
}
